// Encoder_40467181863002
// MI455X (gfx1250) — compile-verified, matrix operand order checked by reading, not run on hardware
//
#include <hip/hip_runtime.h>
#include <math.h>

#define BB 64
#define TT 512
#define EE 1024
#define UU 1024
#define NU3 3072

#define KCHUNK  128            // K depth staged per async stage
#define BSTRIDE 136            // 128 + 8 bf16 pad -> 272B pitch, bank-conflict-free

typedef __attribute__((ext_vector_type(16))) __bf16 v16bf;
typedef __attribute__((ext_vector_type(8)))  __bf16 v8bf;
typedef __attribute__((ext_vector_type(8)))  float  v8f;

// ---------------------------------------------------------------------------
// WMMA helper: D = A(16x32 bf16) x B(32x16 bf16) + C(16x16 f32)
// ---------------------------------------------------------------------------
__device__ __forceinline__ v8f wmma_bf16(v16bf a, v16bf b, v8f c) {
    return __builtin_amdgcn_wmma_f32_16x16x32_bf16(
        /*neg_a=*/false, a, /*neg_b=*/false, b,
        /*c_mod=*/(short)0, c, /*reuse_a=*/false, /*reuse_b=*/false);
}

// CDNA5 async copy: 16 bytes per lane, global -> LDS, tracked by ASYNCcnt.
// VDST = LDS byte address (low 32 bits of generic shared pointer), VADDR = u64.
__device__ __forceinline__ void async_copy_b128(const void* gptr, void* lptr) {
    unsigned lds_off = (unsigned)(size_t)lptr;
    asm volatile("global_load_async_to_lds_b128 %0, %1, off"
                 :: "v"(lds_off), "v"(gptr) : "memory");
}
__device__ __forceinline__ void wait_async0() {
    asm volatile("s_wait_asynccnt 0x0" ::: "memory");
}

// Fragment loader for bf16 rows laid out contiguously in K.
// ISA 16-bit A/B 16x32 layout: lanes 0-15 hold K {kb..kb+7, kb+16..kb+23}
// with kb = 0, lanes 16-31 with kb = 8. Ptr is pre-offset by (k0 + kbase).
// Works for both global (global_load_b128) and LDS (ds_load_b128) sources.
__device__ __forceinline__ v16bf load_frag_bf16(const __bf16* p) {
    v8bf lo = *(const v8bf*)(p);
    v8bf hi = *(const v8bf*)(p + 16);
    v16bf r;
#pragma unroll
    for (int i = 0; i < 8; ++i) { r[i] = lo[i]; r[i + 8] = hi[i]; }
    return r;
}

// Same fragment shape, fp32 source (embedding rows): convert in-regs.
__device__ __forceinline__ v16bf load_frag_f32(const float* p) {
    float4 x0 = *(const float4*)(p);
    float4 x1 = *(const float4*)(p + 4);
    float4 x2 = *(const float4*)(p + 16);
    float4 x3 = *(const float4*)(p + 20);
    v16bf r;
    r[0]  = (__bf16)x0.x; r[1]  = (__bf16)x0.y; r[2]  = (__bf16)x0.z; r[3]  = (__bf16)x0.w;
    r[4]  = (__bf16)x1.x; r[5]  = (__bf16)x1.y; r[6]  = (__bf16)x1.z; r[7]  = (__bf16)x1.w;
    r[8]  = (__bf16)x2.x; r[9]  = (__bf16)x2.y; r[10] = (__bf16)x2.z; r[11] = (__bf16)x2.w;
    r[12] = (__bf16)x3.x; r[13] = (__bf16)x3.y; r[14] = (__bf16)x3.z; r[15] = (__bf16)x3.w;
    return r;
}

// ---------------------------------------------------------------------------
// Kernel 0: transpose+convert Wx[E,3U] and Wh[U,3U] (fp32) -> WxT/WhT[3U,K] bf16
// ---------------------------------------------------------------------------
__global__ __launch_bounds__(256) void convert_weights_kernel(
    const float* __restrict__ Wx, const float* __restrict__ Wh,
    __bf16* __restrict__ WxT, __bf16* __restrict__ WhT)
{
    const long total = (long)NU3 * EE;
    long i = (long)blockIdx.x * 256 + threadIdx.x;
    bool second = (i >= total);
    long o = second ? (i - total) : i;
    int n = (int)(o >> 10);
    int k = (int)(o & 1023);
    float v = second ? Wh[(long)k * NU3 + n] : Wx[(long)k * NU3 + n];
    if (second) WhT[o] = (__bf16)v; else WxT[o] = (__bf16)v;
}

// ---------------------------------------------------------------------------
// Kernel 1: XP[b*T+t, 3U] = emb[x[b,t]] @ Wx + b_i   (bf16 WMMA, f32 accum)
// Grid: (3072/64, 32768/128). Block: 256 (8 waves). Wave w owns M-tile
// m0 = by*128 + w*16; the 64-column B tile is shared by all 8 waves, so it is
// staged through LDS with double-buffered async-to-LDS copies (8 stages of
// K=128). Per k-step: batch-load all 4 B fragments, then issue 4 WMMAs
// back-to-back (one dscnt wait per k-step instead of per tile).
// ---------------------------------------------------------------------------
__global__ __launch_bounds__(256) void xp_gemm_kernel(
    const int*   __restrict__ x,    const float* __restrict__ emb,
    const __bf16* __restrict__ WxT, const float* __restrict__ b_i,
    float* __restrict__ XP)
{
    __shared__ __align__(16) __bf16 Bs[2][64 * BSTRIDE];   // 2 x 17 KB

    const int tid     = threadIdx.x;
    const int lane    = tid & 31;
    const int wave    = tid >> 5;
    const int lidx    = lane & 15;
    const int halfsel = lane >> 4;
    const int kbase   = halfsel * 8;

    const int m0 = blockIdx.y * 128 + wave * 16;
    const int n0 = blockIdx.x * 64;

    const int row   = m0 + lidx;                 // flat b*T + t, < 32768
    const int token = x[row];
    const float* arow = emb + (long)token * EE + kbase;

    // async staging: thread -> (col, quarter); 4 x b128 per thread per stage
    const int scol = tid >> 2;                   // 0..63
    const int sq   = tid & 3;                    // 0..3
    const __bf16* gB = WxT + (long)(n0 + scol) * EE;

    auto issue_stage = [&](int k0, int buf) {
#pragma unroll
        for (int i = 0; i < 4; ++i) {
            const int kk = (sq + 4 * i) * 8;     // 0..120 step 8 (quarter-strided)
            async_copy_b128(gB + k0 + kk, &Bs[buf][scol * BSTRIDE + kk]);
        }
    };

    v8f acc0 = {}, acc1 = {}, acc2 = {}, acc3 = {};

    issue_stage(0, 0);
    for (int s = 0; s < EE / KCHUNK; ++s) {
        wait_async0();          // my stage-s copies have landed in LDS
        __syncthreads();        // everyone's stage-s data ready; stage s-1 reads done
        if (s + 1 < EE / KCHUNK) issue_stage((s + 1) * KCHUNK, (s + 1) & 1);

        const __bf16* bs = &Bs[s & 1][0];
#pragma unroll
        for (int kl = 0; kl < KCHUNK; kl += 32) {
            v16bf a  = load_frag_f32(arow + s * KCHUNK + kl);
            const __bf16* bcol = bs + kl + kbase;
            v16bf b0 = load_frag_bf16(bcol + (lidx +  0) * BSTRIDE);
            v16bf b1 = load_frag_bf16(bcol + (lidx + 16) * BSTRIDE);
            v16bf b2 = load_frag_bf16(bcol + (lidx + 32) * BSTRIDE);
            v16bf b3 = load_frag_bf16(bcol + (lidx + 48) * BSTRIDE);
            acc0 = wmma_bf16(a, b0, acc0);
            acc1 = wmma_bf16(a, b1, acc1);
            acc2 = wmma_bf16(a, b2, acc2);
            acc3 = wmma_bf16(a, b3, acc3);
        }
    }

#pragma unroll
    for (int v = 0; v < 8; ++v) {
        const int  r    = m0 + v + halfsel * 8;
        const long base = (long)r * NU3;
        const int  c0   = n0 + lidx;
        XP[base + c0 +  0] = acc0[v] + b_i[c0 +  0];
        XP[base + c0 + 16] = acc1[v] + b_i[c0 + 16];
        XP[base + c0 + 32] = acc2[v] + b_i[c0 + 32];
        XP[base + c0 + 48] = acc3[v] + b_i[c0 + 48];
    }
}

// ---------------------------------------------------------------------------
// Grid-wide barrier (persistent scan kernel, 16 resident WGs).
// ---------------------------------------------------------------------------
__device__ __forceinline__ void grid_barrier(unsigned* cnt, unsigned target) {
    __syncthreads();
    if (threadIdx.x == 0) {
        __hip_atomic_fetch_add(cnt, 1u, __ATOMIC_RELEASE, __HIP_MEMORY_SCOPE_AGENT);
    }
    while (__hip_atomic_load(cnt, __ATOMIC_ACQUIRE, __HIP_MEMORY_SCOPE_AGENT) < target) {
        __builtin_amdgcn_s_sleep(2);
    }
    __syncthreads();
}

__device__ __forceinline__ float sigmoidf_(float v) {
    return 1.0f / (1.0f + __expf(-v));
}

// ---------------------------------------------------------------------------
// Kernel 2: persistent GRU scan. Grid = 16 WGs x 256 threads.
// WG g owns units u0 = g*64 (gate cols u0, u0+1024, u0+2048 of h@Wh).
// Wave w: M-tile (batch rows) m0=(w&3)*16, column blocks {2*(w>>2), +1}.
// h state: f32 in registers (each (b,u) owned by one thread); bf16 ping-pong
// buffers in ws feed the next step's WMMA A operand. Per k-step all 6 B
// fragments are batch-loaded before the 6 WMMAs.
// ---------------------------------------------------------------------------
__global__ __launch_bounds__(256) void gru_scan_kernel(
    const int*   __restrict__ x,   const float* __restrict__ hidden,
    const __bf16* __restrict__ WhT, const float* __restrict__ b_r,
    const float* __restrict__ XP,
    __bf16* hB0, __bf16* hB1, unsigned* cnt, float* __restrict__ out)
{
    const int lane    = threadIdx.x & 31;
    const int wave    = threadIdx.x >> 5;
    const int lidx    = lane & 15;
    const int halfsel = lane >> 4;
    const int kbase   = halfsel * 8;

    const int u0  = blockIdx.x * 64;
    const int m0  = (wave & 3) * 16;
    const int jb0 = (wave >> 2) * 2;

    float hreg[2][8];
#pragma unroll
    for (int j = 0; j < 2; ++j) {
        const int u = u0 + (jb0 + j) * 16 + lidx;
#pragma unroll
        for (int v = 0; v < 8; ++v) {
            const int b = m0 + v + halfsel * 8;
            float h0 = hidden[b * UU + u];
            hreg[j][v] = h0;
            hB0[b * UU + u] = (__bf16)h0;
        }
    }

    const __bf16* bptr[2][3];
#pragma unroll
    for (int j = 0; j < 2; ++j)
#pragma unroll
        for (int g = 0; g < 3; ++g)
            bptr[j][g] = WhT + (long)(g * UU + u0 + (jb0 + j) * 16 + lidx) * UU + kbase;

    unsigned target = gridDim.x;
    grid_barrier(cnt, target);

    for (int t = 0; t < TT; ++t) {
        const __bf16* hc = (t & 1) ? hB1 : hB0;
        __bf16*       hn = (t & 1) ? hB0 : hB1;

        v8f acc[2][3] = {};
        const __bf16* arow = hc + (m0 + lidx) * UU + kbase;
        for (int k0 = 0; k0 < UU; k0 += 32) {
            v16bf a = load_frag_bf16(arow + k0);
            v16bf bf[2][3];
#pragma unroll
            for (int j = 0; j < 2; ++j)
#pragma unroll
                for (int g = 0; g < 3; ++g)
                    bf[j][g] = load_frag_bf16(bptr[j][g] + k0);
#pragma unroll
            for (int j = 0; j < 2; ++j)
#pragma unroll
                for (int g = 0; g < 3; ++g)
                    acc[j][g] = wmma_bf16(a, bf[j][g], acc[j][g]);
        }

#pragma unroll
        for (int j = 0; j < 2; ++j) {
            const int u = u0 + (jb0 + j) * 16 + lidx;
#pragma unroll
            for (int v = 0; v < 8; ++v) {
                const int  b   = m0 + v + halfsel * 8;
                const long row = (long)b * TT + t;
                const long xb  = row * NU3;
                float xz = XP[xb + u];
                float xr = XP[xb + u + UU];
                float xh = XP[xb + u + 2 * UU];
                float hz = acc[j][0][v] + b_r[u];
                float hr = acc[j][1][v] + b_r[u + UU];
                float hh = acc[j][2][v] + b_r[u + 2 * UU];
                float z    = sigmoidf_(xz + hz);
                float r    = sigmoidf_(xr + hr);
                float cand = tanhf(xh + r * hh);
                float hold = hreg[j][v];
                float hnew = z * hold + (1.0f - z) * cand;
                hnew = (x[row] != 0) ? hnew : hold;      // mask_zero copy-through
                hreg[j][v] = hnew;
                out[row * UU + u] = hnew;
                hn[b * UU + u] = (__bf16)hnew;
            }
        }

        target += gridDim.x;
        grid_barrier(cnt, target);
    }

    float* state = out + (long)BB * TT * UU;
#pragma unroll
    for (int j = 0; j < 2; ++j) {
        const int u = u0 + (jb0 + j) * 16 + lidx;
#pragma unroll
        for (int v = 0; v < 8; ++v) {
            const int b = m0 + v + halfsel * 8;
            state[b * UU + u] = hreg[j][v];
        }
    }
}

// ---------------------------------------------------------------------------
// Host side
// ---------------------------------------------------------------------------
extern "C" void kernel_launch(void* const* d_in, const int* in_sizes, int n_in,
                              void* d_out, int out_size, void* d_ws, size_t ws_size,
                              hipStream_t stream) {
    (void)in_sizes; (void)n_in; (void)out_size; (void)ws_size;

    const int*   x      = (const int*)  d_in[0];   // [B,T]
    const float* hidden = (const float*)d_in[1];   // [B,U]
    const float* emb    = (const float*)d_in[2];   // [V,E]
    const float* Wx     = (const float*)d_in[3];   // [E,3U]
    const float* Wh     = (const float*)d_in[4];   // [U,3U]
    const float* b_i    = (const float*)d_in[5];   // [3U]
    const float* b_r    = (const float*)d_in[6];   // [3U]
    float*       out    = (float*)d_out;           // [B,T,U] ++ [B,U]

    char* w = (char*)d_ws;
    __bf16*   WxT = (__bf16*)(w);                        // 6291456 B
    __bf16*   WhT = (__bf16*)(w + 6291456);              // 6291456 B
    __bf16*   hB0 = (__bf16*)(w + 12582912);             // 131072 B
    __bf16*   hB1 = (__bf16*)(w + 12713984);             // 131072 B
    unsigned* cnt = (unsigned*)(w + 12845056);           // 256 B
    float*    XP  = (float*)(w + 12845312);              // 402653184 B

    hipMemsetAsync(cnt, 0, 256, stream);

    convert_weights_kernel<<<2 * NU3 * EE / 256, 256, 0, stream>>>(Wx, Wh, WxT, WhT);

    dim3 g1(NU3 / 64, (BB * TT) / 128);
    xp_gemm_kernel<<<g1, 256, 0, stream>>>(x, emb, WxT, b_i, XP);

    gru_scan_kernel<<<UU / 64, 256, 0, stream>>>(x, hidden, WhT, b_r, XP,
                                                 hB0, hB1, cnt, out);
}
